// SpecialTransformerLayer_87119116632179
// MI455X (gfx1250) — matrix operand order repaired and checked by reading, not compiled, not run on hardware
//
#include <hip/hip_runtime.h>
#include <math.h>

typedef __attribute__((ext_vector_type(16))) _Float16 v16h;
typedef __attribute__((ext_vector_type(8)))  float    v8f;

#define INV_SQRT2 0.70710678118654752440f

// ---------------------------------------------------------------------------
// GroupNorm: one block per (batch, group). C=256, groups=32 -> 8 ch/group.
// ---------------------------------------------------------------------------
__global__ void gn_kernel(const float* __restrict__ x,
                          const float* __restrict__ scale,
                          const float* __restrict__ bias,
                          float* __restrict__ out,
                          int C, int P, int groups) {
  const int b = blockIdx.y;
  const int g = blockIdx.x;
  const int cpg = C / groups;
  const int n = cpg * P;
  const size_t base = ((size_t)b * C + (size_t)g * cpg) * (size_t)P;
  const float* xg = x + base;
  float* og = out + base;

  float s = 0.f, s2 = 0.f;
  for (int i = threadIdx.x; i < n; i += blockDim.x) {
    float v = xg[i];
    s += v; s2 += v * v;
  }
  __shared__ float sh0[256], sh1[256];
  sh0[threadIdx.x] = s; sh1[threadIdx.x] = s2;
  __syncthreads();
  for (int off = 128; off > 0; off >>= 1) {
    if ((int)threadIdx.x < off) {
      sh0[threadIdx.x] += sh0[threadIdx.x + off];
      sh1[threadIdx.x] += sh1[threadIdx.x + off];
    }
    __syncthreads();
  }
  const float mean = sh0[0] / (float)n;
  const float var  = sh1[0] / (float)n - mean * mean;
  const float rstd = rsqrtf(var + 1e-6f);
  for (int i = threadIdx.x; i < n; i += blockDim.x) {
    int c = g * cpg + i / P;
    og[i] = (xg[i] - mean) * rstd * scale[c] + bias[c];
  }
}

// ---------------------------------------------------------------------------
// Channel linear via WMMA: Y[b,d,p] = bias[d] + sum_c X[b,c,p]*W[c,d]
// X: (B, Cin, P) channel-major, W: (Cin, Dout) row-major, Y: (B, Dout, P).
// Block = 4 waves = 16 positions x 64 out-channels. The 32x16 activation
// slab is staged to LDS ONCE per K-step (coalesced float4 loads, scattered
// into A-fragment order), double-buffered; all 4 waves consume it from LDS
// as a single contiguous 32B read per lane. 4x less activation traffic.
// ---------------------------------------------------------------------------
__global__ void linc_wmma_kernel(const float* __restrict__ X,
                                 const float* __restrict__ W,
                                 const float* __restrict__ bias,
                                 float* __restrict__ Y,
                                 int Cin, int Dout, int P) {
  const int tid  = threadIdx.x;        // 0..127
  const int lane = tid & 31;
  const int wv   = tid >> 5;           // wave id 0..3
  const int hl   = lane >> 4;          // lane half
  const int l16  = lane & 15;
  const int pbase = blockIdx.x * 16;
  const int nbase = blockIdx.y * 64 + wv * 16;
  const int b     = blockIdx.z;

  const float* Xb = X + (size_t)b * (size_t)Cin * (size_t)P;

  // Double-buffered A slabs, pre-swizzled into WMMA A-fragment order:
  // element i of lane l lives at [l*16 + i].
  __shared__ alignas(32) _Float16 asmem[2][32 * 16];

  const float bval = bias[nbase + l16];
  v8f acc;
#pragma unroll
  for (int r = 0; r < 8; ++r) acc[r] = bval;

  // Staging role of this thread: one float4 = row srow (k_local), cols sseg*4..+3
  const int srow = tid >> 2;                         // k_local 0..31
  const int sseg = tid & 3;
  const int shl  = (srow >> 3) & 1;                  // dest lane half
  const int sii  = (srow & 7) + ((srow & 16) ? 8 : 0); // dest fragment element

  const int nk = Cin / 32;
  for (int ks = 0; ks < nk; ++ks) {
    const int k0  = ks * 32;
    const int buf = ks & 1;
    if (k0 + 32 < Cin)
      __builtin_prefetch(&W[(size_t)(k0 + 32) * Dout + nbase], 0, 1);

    // ---- stage X slab (32 k x 16 p) into LDS in A-fragment order ----
    const float4 xv =
        *(const float4*)&Xb[(size_t)(k0 + srow) * P + pbase + sseg * 4];
    {
      _Float16* dst = &asmem[buf][0];
      float vs[4] = {xv.x, xv.y, xv.z, xv.w};
#pragma unroll
      for (int j = 0; j < 4; ++j) {
        int m = sseg * 4 + j;                        // position within tile
        dst[(m + 16 * shl) * 16 + sii] = (_Float16)vs[j];
      }
    }
    __syncthreads();

    // ---- A fragment: one contiguous 32B LDS read, shared by all waves ----
    v16h a = *(const v16h*)&asmem[buf][lane * 16];

    // ---- B fragment from global (lane-coalesced along Dout) ----
    v16h bm;
#pragma unroll
    for (int i = 0; i < 16; ++i) {
      int kb = (i < 8 ? i : i + 8) + hl * 8;   // B layout: table order, matches A
      bm[i] = (_Float16)W[(size_t)(k0 + kb) * Dout + nbase + l16];
    }
    acc = __builtin_amdgcn_wmma_f32_16x16x32_f16(false, a, false, bm,
                                                 (short)0, acc, false, false);
  }

  float* Yb = Y + (size_t)b * (size_t)Dout * (size_t)P;
#pragma unroll
  for (int r = 0; r < 8; ++r) {
    int p = pbase + r + 8 * hl;   // C layout: lanes0-15 M=0..7, lanes16-31 M=8..15
    Yb[(size_t)(nbase + l16) * P + p] = acc[r];
  }
}

// ---------------------------------------------------------------------------
// Multi-head attention (flash-style, online softmax), one wave per
// (batch, head, 16-query tile).  Q:(B,AD,P2)  K,V:(B,AD,P)  O:(B,AD,P2)
// DH = head dim (16 for cross-attn, 64 for self-attn); AD = 4*DH.
// ---------------------------------------------------------------------------
template <int DH>
__global__ void attend_wmma_kernel(const float* __restrict__ Q,
                                   const float* __restrict__ K,
                                   const float* __restrict__ V,
                                   float* __restrict__ O,
                                   int P2, int P, float scale) {
  constexpr int NDT = DH / 16;            // output dim tiles
  constexpr int NKS = (DH + 31) / 32;     // K-steps for score GEMM
  const int lane = threadIdx.x;
  const int hl   = lane >> 4;
  const int l16  = lane & 15;
  const int qbase = blockIdx.x * 16;
  const int h     = blockIdx.y;
  const int b     = blockIdx.z;
  const int AD    = 4 * DH;

  const float* Qb = Q + ((size_t)b * AD + (size_t)h * DH) * (size_t)P2;
  const float* Kb = K + ((size_t)b * AD + (size_t)h * DH) * (size_t)P;
  const float* Vb = V + ((size_t)b * AD + (size_t)h * DH) * (size_t)P;

  // Preload Q tile as A fragments (zero-padded when DH==16).
  v16h qa[NKS];
#pragma unroll
  for (int f = 0; f < NKS; ++f) {
#pragma unroll
    for (int i = 0; i < 16; ++i) {
      int kl = (i < 8 ? i : i + 8) + hl * 8 + f * 32;
      qa[f][i] = (kl < DH) ? (_Float16)Qb[(size_t)kl * P2 + qbase + l16]
                           : (_Float16)0.0f;
    }
  }

  v8f ov[NDT];
#pragma unroll
  for (int t = 0; t < NDT; ++t)
#pragma unroll
    for (int r = 0; r < 8; ++r) ov[t][r] = 0.f;

  float mrow[8], lrow[8];
#pragma unroll
  for (int r = 0; r < 8; ++r) { mrow[r] = -1.0e30f; lrow[r] = 0.f; }

  __shared__ _Float16 pshare[16 * 16];

  const int nkt = P / 16;
  for (int kt = 0; kt < nkt; ++kt) {
    const int kbase = kt * 16;
    // ---- scores S = Q^T K (16 queries x 16 keys) ----
    v8f s;
#pragma unroll
    for (int r = 0; r < 8; ++r) s[r] = 0.f;
#pragma unroll
    for (int ks = 0; ks < NKS; ++ks) {
      v16h kb;
#pragma unroll
      for (int i = 0; i < 16; ++i) {
        int kr = (i < 8 ? i : i + 8) + hl * 8 + ks * 32;
        kb[i] = (kr < DH) ? (_Float16)Kb[(size_t)kr * P + kbase + l16]
                          : (_Float16)0.0f;
      }
      s = __builtin_amdgcn_wmma_f32_16x16x32_f16(false, qa[ks], false, kb,
                                                 (short)0, s, false, false);
    }
    // ---- online softmax: rows live across 16-lane half-groups ----
    float alpha[8];
#pragma unroll
    for (int r = 0; r < 8; ++r) {
      float sv = s[r] * scale;
      float rmax = sv;
      rmax = fmaxf(rmax, __shfl_xor(rmax, 1, 32));
      rmax = fmaxf(rmax, __shfl_xor(rmax, 2, 32));
      rmax = fmaxf(rmax, __shfl_xor(rmax, 4, 32));
      rmax = fmaxf(rmax, __shfl_xor(rmax, 8, 32));
      float mnew = fmaxf(mrow[r], rmax);
      alpha[r] = __expf(mrow[r] - mnew);
      float p = __expf(sv - mnew);
      float rs = p;
      rs += __shfl_xor(rs, 1, 32);
      rs += __shfl_xor(rs, 2, 32);
      rs += __shfl_xor(rs, 4, 32);
      rs += __shfl_xor(rs, 8, 32);
      lrow[r] = lrow[r] * alpha[r] + rs;
      mrow[r] = mnew;
      s[r] = p;
    }
    // ---- restage P (C layout) -> A layout via LDS ----
    __syncthreads();
#pragma unroll
    for (int r = 0; r < 8; ++r)
      pshare[(r + 8 * hl) * 16 + l16] = (_Float16)s[r];
    __syncthreads();
    v16h pa;
#pragma unroll
    for (int i = 0; i < 16; ++i) {
      int kl = (i < 8 ? i : i + 8) + hl * 8;   // 0..31; only K<16 valid
      pa[i] = (kl < 16) ? pshare[l16 * 16 + kl] : (_Float16)0.0f;
    }
    __syncthreads();
    // ---- O += P * V^T per 16-dim tile ----
#pragma unroll
    for (int t = 0; t < NDT; ++t) {
      v16h vb;
#pragma unroll
      for (int i = 0; i < 16; ++i) {
        int kr = (i < 8 ? i : i + 8) + hl * 8; // key index within tile
        vb[i] = (kr < 16)
                    ? (_Float16)Vb[(size_t)(t * 16 + l16) * P + kbase + kr]
                    : (_Float16)0.0f;
      }
#pragma unroll
      for (int r = 0; r < 8; ++r) ov[t][r] *= alpha[r];
      ov[t] = __builtin_amdgcn_wmma_f32_16x16x32_f16(false, pa, false, vb,
                                                     (short)0, ov[t], false, false);
    }
  }

  float* Ob = O + ((size_t)b * AD + (size_t)h * DH) * (size_t)P2;
#pragma unroll
  for (int t = 0; t < NDT; ++t)
#pragma unroll
    for (int r = 0; r < 8; ++r)
      Ob[(size_t)(t * 16 + l16) * P2 + qbase + r + 8 * hl] = ov[t][r] / lrow[r];
}

// ---------------------------------------------------------------------------
// Elementwise helpers
// ---------------------------------------------------------------------------
__global__ void resid_kernel(const float* __restrict__ a,
                             const float* __restrict__ h,
                             float* __restrict__ out, size_t n) {
  size_t i = (size_t)blockIdx.x * blockDim.x + threadIdx.x;
  if (i < n) out[i] = (a[i] + h[i]) * INV_SQRT2;
}

__global__ void gelu_kernel(float* __restrict__ x, size_t n) {
  size_t i = (size_t)blockIdx.x * blockDim.x + threadIdx.x;
  if (i < n) {
    float v = x[i];
    x[i] = 0.5f * v * (1.0f + erff(v * INV_SQRT2));
  }
}

// c[b,d,:] += stepemb[b,:] @ Ws[:,d] + bs[d]
__global__ void step_add_kernel(const float* __restrict__ stepemb,
                                const float* __restrict__ Ws,
                                const float* __restrict__ bs,
                                float* __restrict__ c, int C, int P) {
  int d = blockIdx.x * blockDim.x + threadIdx.x;
  int b = blockIdx.y;
  if (d >= C) return;
  const float* se = stepemb + (size_t)b * C;
  float s = bs[d];
  for (int k = 0; k < C; ++k) s += se[k] * Ws[(size_t)k * C + d];
  float* cp = c + ((size_t)b * C + d) * (size_t)P;
  for (int p = 0; p < P; ++p) cp[p] += s;
}

// ---------------------------------------------------------------------------
// Host orchestration
// ---------------------------------------------------------------------------
static void gn(const float* in, const float* s, const float* bsc, float* out,
               int P, hipStream_t st) {
  dim3 g(32, 16);
  gn_kernel<<<g, 256, 0, st>>>(in, s, bsc, out, 256, P, 32);
}
static void linc(const float* X, const float* W, const float* bias, float* Y,
                 int Cin, int Dout, int P, hipStream_t st) {
  dim3 g(P / 16, Dout / 64, 16);
  linc_wmma_kernel<<<g, 128, 0, st>>>(X, W, bias, Y, Cin, Dout, P);
}
static void attend(const float* Q, const float* K, const float* V, float* O,
                   int AD, int P2, int P, hipStream_t st) {
  dim3 g(P2 / 16, 4, 16);
  float scale = 1.0f / sqrtf((float)AD);
  if (AD == 64)
    attend_wmma_kernel<16><<<g, 32, 0, st>>>(Q, K, V, O, P2, P, scale);
  else
    attend_wmma_kernel<64><<<g, 32, 0, st>>>(Q, K, V, O, P2, P, scale);
}
static void resid(const float* a, const float* h, float* out, size_t n,
                  hipStream_t st) {
  resid_kernel<<<(unsigned)((n + 255) / 256), 256, 0, st>>>(a, h, out, n);
}
static void gelu(float* x, size_t n, hipStream_t st) {
  gelu_kernel<<<(unsigned)((n + 255) / 256), 256, 0, st>>>(x, n);
}

extern "C" void kernel_launch(void* const* d_in, const int* in_sizes, int n_in,
                              void* d_out, int out_size, void* d_ws, size_t ws_size,
                              hipStream_t stream) {
  (void)in_sizes; (void)n_in; (void)out_size; (void)ws_size;
  const float* x       = (const float*)d_in[0];   // (16,256,32,32)
  const float* cells   = (const float*)d_in[1];   // (16,256,16,16)
  const float* stepemb = (const float*)d_in[2];   // (16,256)
  const float* Pp[64];
  for (int i = 0; i < 51; ++i) Pp[i] = (const float*)d_in[i];

  const int PX = 1024, PC = 256;
  const size_t NX  = (size_t)16 * 256 * PX;   // 4,194,304
  const size_t NC  = (size_t)16 * 256 * PC;   // 1,048,576
  const size_t NFC = (size_t)16 * 512 * PC;   // 2,097,152
  const size_t NFX = (size_t)16 * 512 * PX;   // 8,388,608

  float* ws = (float*)d_ws;
  float* A0 = ws;                 // x-sized
  float* A1 = A0 + NX;            // x-sized
  float* A2 = A1 + NX;            // x-sized
  float* A3 = A2 + NX;            // FFx-sized
  float* B0 = A3 + NFX;           // cells-sized
  float* B1 = B0 + NC;
  float* B2 = B1 + NC;
  float* B3 = B2 + NC;
  float* B4 = B3 + NC;
  float* B5 = B4 + NC;
  float* FC = B5 + NC;            // FFc-sized

  float* x_out = (float*)d_out;          // first NX floats
  float* c_out = x_out + NX;             // next NC floats

  // ===================== ca1: pull x into cells =====================
  gn(x,     Pp[3], Pp[4], A0, PX, stream);               // an  = gnA(x)
  gn(cells, Pp[5], Pp[6], B0, PC, stream);               // bn  = gnB(cells)
  linc(B0, Pp[7],  Pp[8],  B1, 256, 64, PC, stream);     // q on cells
  linc(A0, Pp[9],  Pp[10], A1, 256, 64, PX, stream);     // k on x
  linc(A0, Pp[11], Pp[12], A2, 256, 64, PX, stream);     // v on x
  attend(B1, A1, A2, B2, 64, PC, PX, stream);            // attn -> (16,64,256)
  linc(B2, Pp[13], Pp[14], B3, 64, 256, PC, stream);     // o-proj
  resid(cells, B3, B4, NC, stream);                      // c = (cells+h)/sqrt2

  // ===================== self-attn on cells =====================
  gn(B4, Pp[15], Pp[16], B0, PC, stream);                // cn
  linc(B0, Pp[17], Pp[18], B1, 256, 256, PC, stream);    // q
  linc(B0, Pp[19], Pp[20], B2, 256, 256, PC, stream);    // k
  linc(B0, Pp[21], Pp[22], B5, 256, 256, PC, stream);    // v
  attend(B1, B2, B5, B0, 256, PC, PC, stream);           // attn (dh=64)
  linc(B0, Pp[23], Pp[24], B1, 256, 256, PC, stream);    // o-proj
  resid(B4, B1, B2, NC, stream);                         // c = (c+h)/sqrt2

  // ===================== step emb + cell FFN =====================
  {
    dim3 g((256 + 63) / 64, 16);
    step_add_kernel<<<g, 64, 0, stream>>>(stepemb, Pp[37], Pp[38], B2, 256, PC);
  }
  gn(B2, Pp[39], Pp[40], B0, PC, stream);                // cn2
  linc(B0, Pp[41], Pp[42], FC, 256, 512, PC, stream);    // ffA_c
  gelu(FC, NFC, stream);
  linc(FC, Pp[43], Pp[44], B1, 512, 256, PC, stream);    // ffB_c
  resid(cells, B1, c_out, NC, stream);                   // cells_out

  // ===================== ca2: pull cells_out into x =====================
  gn(c_out, Pp[25], Pp[26], B0, PC, stream);             // an2 = gnA(cells_out)
  gn(x,     Pp[27], Pp[28], A0, PX, stream);             // xn2 = gnB(x)
  linc(A0, Pp[29], Pp[30], A1, 256, 64, PX, stream);     // q on x
  linc(B0, Pp[31], Pp[32], B1, 256, 64, PC, stream);     // k on cells_out
  linc(B0, Pp[33], Pp[34], B3, 256, 64, PC, stream);     // v on cells_out
  attend(A1, B1, B3, A2, 64, PX, PC, stream);            // attn -> (16,64,1024)
  linc(A2, Pp[35], Pp[36], A0, 64, 256, PX, stream);     // o-proj
  resid(x, A0, A1, NX, stream);                          // x2 = (x+h)/sqrt2

  // ===================== x FFN =====================
  gn(A1, Pp[45], Pp[46], A0, PX, stream);                // xn3
  linc(A0, Pp[47], Pp[48], A3, 256, 512, PX, stream);    // ffA_x
  gelu(A3, NFX, stream);
  linc(A3, Pp[49], Pp[50], A2, 512, 256, PX, stream);    // ffB_x
  resid(A1, A2, x_out, NX, stream);                      // x_out
}
